// HoiOutputLayers_20212116095223
// MI455X (gfx1250) — compile-verified, matrix operand order checked by reading, not run on hardware
//
#include <hip/hip_runtime.h>
#include <hip/hip_bf16.h>

typedef __attribute__((ext_vector_type(16))) __bf16 v16bf;
typedef __attribute__((ext_vector_type(8)))  __bf16 v8bf;
typedef __attribute__((ext_vector_type(8)))  float  v8f;
typedef __attribute__((ext_vector_type(4)))  unsigned int v4u;
typedef __attribute__((ext_vector_type(8)))  int v8i;
typedef __attribute__((ext_vector_type(4)))  int v4i;

#define RROWS 32768
#define K1    3072
#define N1    1024
#define N2    50
#define N2P   64

// LDS row stride for 32-element bf16 tile rows, padded to 40 bf16 (20 dwords):
// 20*L mod 64 for L=0..15 covers all 64 banks in disjoint 4-dword spans.
#define AROW  40

// ---------------- prepack: W1 [K1][N1] f32 -> w1t [N1][K1] bf16 ----------------
__global__ __launch_bounds__(256) void pack_w1t(const float* __restrict__ w1,
                                                __bf16* __restrict__ w1t) {
    int idx = blockIdx.x * 256 + threadIdx.x;
    int k = idx / N1;
    int n = idx % N1;
    w1t[(size_t)n * K1 + k] = (__bf16)w1[idx];
}

// ---------------- prepack: W2 [N1][N2] f32 -> w2t [N2P][N1] bf16 (zero pad) ----
__global__ __launch_bounds__(256) void pack_w2t(const float* __restrict__ w2,
                                                __bf16* __restrict__ w2t) {
    int idx = blockIdx.x * 256 + threadIdx.x;
    int k = idx / N2P;
    int n = idx % N2P;
    float v = (n < N2) ? w2[(size_t)k * N2 + n] : 0.0f;
    w2t[(size_t)n * N1 + k] = (__bf16)v;
}

// ---------------- TDM: 2D bf16 tile load -> LDS with row padding ---------------
// Loads a tile_w x tile_h bf16 tile; pads each 16-dword row by 4 dwords so the
// LDS row stride is 20 dwords (40 bf16) -> conflict-free b128 fragment reads.
__device__ __forceinline__ void tdm_load_w(unsigned lds_off, const void* gaddr,
                                           unsigned tensor_w, unsigned tensor_h,
                                           unsigned tile_w, unsigned tile_h,
                                           unsigned stride_elems) {
    unsigned long long ga = (unsigned long long)(uintptr_t)gaddr;
    v4u g0;
    g0.x = 1u;                                              // count=1 (valid user D#)
    g0.y = lds_off;                                         // LDS byte address
    g0.z = (unsigned)(ga & 0xFFFFFFFFu);                    // global_addr[31:0]
    g0.w = (unsigned)((ga >> 32) & 0x01FFFFFFu)             // global_addr[56:32]
         | (2u << 30);                                      // type=2 ("image")
    v8i g1;
    g1[0] = (int)((1u << 16)                                // data_size = 2 bytes
         | (1u << 20)                                       // pad_enable
         | (3u << 22)                                       // pad_interval: 16 dwords
         | (3u << 25));                                     // pad_amount: 4 dwords
    g1[1] = (int)((tensor_w & 0xFFFFu) << 16);              // tensor_dim0[15:0]
    g1[2] = (int)((tensor_w >> 16) | ((tensor_h & 0xFFFFu) << 16)); // dim0 hi | dim1 lo
    g1[3] = (int)((tensor_h >> 16) | (tile_w << 16));       // dim1 hi | tile_dim0
    g1[4] = (int)tile_h;                                    // tile_dim1 (tile_dim2=0)
    g1[5] = (int)stride_elems;                              // tensor_dim0_stride[31:0]
    g1[6] = 0;
    g1[7] = 0;
    v4i g2 = {0, 0, 0, 0};
    v4i g3 = {0, 0, 0, 0};
    v8i g4 = {0, 0, 0, 0, 0, 0, 0, 0};
    __builtin_amdgcn_tensor_load_to_lds(g0, g1, g2, g3, g4, 0);
}

// ---------------- GEMM1: h = relu(x @ W1 + b1), h stored bf16 ------------------
// grid: (N1/128, RROWS/128), block 256 = 8 waves (2M x 4N); wave tile 64x32.
// x tile: prefetched f32 -> regs -> cvt once -> bf16 LDS (staged, overlapped).
// w1t tile: TDM double-buffered into padded LDS. One barrier per K-step.
__global__ __launch_bounds__(256) void gemm1_relu_tdm(const float*  __restrict__ x,
                                                      const __bf16* __restrict__ w1t,
                                                      const float*  __restrict__ b1,
                                                      __bf16*       __restrict__ h) {
    __shared__ __bf16 ldsA[2][128 * AROW];   // 2 x 10 KB  (x tile, bf16)
    __shared__ __bf16 ldsW[2][128 * AROW];   // 2 x 10 KB  (w1t tile, bf16)

    const int lane  = threadIdx.x & 31;
    const int wave  = threadIdx.x >> 5;
    const int waveM = wave >> 2;                 // 0..1
    const int waveN = wave & 3;                  // 0..3
    const int rowBlk = blockIdx.y * 128;
    const int colBlk = blockIdx.x * 128;
    const int l15  = lane & 15;
    const int koff = (lane >> 4) << 3;           // 0 or 8 (16-bit A/B lane K layout)

    // per-thread staging slot: thread t owns row t/2, k-half (t&1)*16
    const int stRow  = threadIdx.x >> 1;
    const int stHalf = (threadIdx.x & 1) * 16;
    const float* xstage = x + (size_t)(rowBlk + stRow) * K1 + stHalf;
    const int stOff = stRow * AROW + stHalf;

    // ---- prologue: fill buffer 0 ----
    if (wave == 0)
        tdm_load_w((unsigned)(uintptr_t)&ldsW[0][0],
                   w1t + (size_t)colBlk * K1, K1, N1, 32u, 128u, K1);
    {
        v8f s0 = *(const v8f*)(xstage);
        v8f s1 = *(const v8f*)(xstage + 8);
        *(v8bf*)&ldsA[0][stOff]     = __builtin_convertvector(s0, v8bf);
        *(v8bf*)&ldsA[0][stOff + 8] = __builtin_convertvector(s1, v8bf);
    }
    if (wave == 0) __builtin_amdgcn_s_wait_tensorcnt(0);
    __syncthreads();

    v8f acc[4][2] = {};

    int ib = 0;
    for (int kb = 0; kb < K1; kb += 32, ib ^= 1) {
        const bool hasNext = (kb + 32 < K1);

        // prefetch next tiles (writes target the buffer read LAST iteration,
        // separated from those reads by the barrier at the end of it)
        v8f s0 = {}, s1 = {};
        if (hasNext) {
            if (wave == 0)
                tdm_load_w((unsigned)(uintptr_t)&ldsW[ib ^ 1][0],
                           w1t + (size_t)colBlk * K1 + (kb + 32), K1, N1, 32u, 128u, K1);
            const float* xp = xstage + kb + 32;
            s0 = *(const v8f*)(xp);
            s1 = *(const v8f*)(xp + 8);
        }

        // ---- compute: pure ds_load + WMMA ----
        v16bf afrag[4];
#pragma unroll
        for (int mt = 0; mt < 4; ++mt) {
            const __bf16* ap = &ldsA[ib][(waveM * 64 + mt * 16 + l15) * AROW + koff];
            v8bf alo = *(const v8bf*)(ap);
            v8bf ahi = *(const v8bf*)(ap + 16);
            afrag[mt] = __builtin_shufflevector(alo, ahi,
                0,1,2,3,4,5,6,7,8,9,10,11,12,13,14,15);
        }
        v16bf bfrag[2];
#pragma unroll
        for (int nt = 0; nt < 2; ++nt) {
            const __bf16* bp = &ldsW[ib][(waveN * 32 + nt * 16 + l15) * AROW + koff];
            v8bf blo = *(const v8bf*)(bp);
            v8bf bhi = *(const v8bf*)(bp + 16);
            bfrag[nt] = __builtin_shufflevector(blo, bhi,
                0,1,2,3,4,5,6,7,8,9,10,11,12,13,14,15);
        }
#pragma unroll
        for (int mt = 0; mt < 4; ++mt)
#pragma unroll
            for (int nt = 0; nt < 2; ++nt)
                acc[mt][nt] = __builtin_amdgcn_wmma_f32_16x16x32_bf16(
                    false, afrag[mt], false, bfrag[nt],
                    (short)0, acc[mt][nt], false, false);

        // convert staged x (fresh regs: no WMMA hazard) and commit to other buffer
        if (hasNext) {
            *(v8bf*)&ldsA[ib ^ 1][stOff]     = __builtin_convertvector(s0, v8bf);
            *(v8bf*)&ldsA[ib ^ 1][stOff + 8] = __builtin_convertvector(s1, v8bf);
        }

        if (wave == 0) __builtin_amdgcn_s_wait_tensorcnt(0);
        __syncthreads();
    }

    // ---- epilogue: bias + relu + bf16 store ----
#pragma unroll
    for (int mt = 0; mt < 4; ++mt) {
#pragma unroll
        for (int nt = 0; nt < 2; ++nt) {
            const int col = colBlk + waveN * 32 + nt * 16 + l15;
            const float bias = b1[col];
#pragma unroll
            for (int v = 0; v < 8; ++v) {
                const int row = rowBlk + waveM * 64 + mt * 16 + ((lane < 16) ? v : v + 8);
                float val = acc[mt][nt][v] + bias;
                val = fmaxf(val, 0.0f);
                h[(size_t)row * N1 + col] = (__bf16)val;
            }
        }
    }
}

// ---------------- GEMM2: out = h @ W2 + b2 (N padded to 64) --------------------
__global__ __launch_bounds__(256) void gemm2_bias(const __bf16* __restrict__ h,
                                                  const __bf16* __restrict__ w2t,
                                                  const float*  __restrict__ b2,
                                                  float*        __restrict__ out) {
    const int lane = threadIdx.x & 31;
    const int wave = threadIdx.x >> 5;
    const int rowBase = blockIdx.x * 128 + wave * 16;
    const int l15  = lane & 15;
    const int koff = (lane >> 4) << 3;

    v8f acc[4] = {};

    for (int kb = 0; kb < N1; kb += 32) {
        const __bf16* ap = h + (size_t)(rowBase + l15) * N1 + kb + koff;
        v8bf alo = *(const v8bf*)(ap);
        v8bf ahi = *(const v8bf*)(ap + 16);
        v16bf afrag = __builtin_shufflevector(alo, ahi,
            0,1,2,3,4,5,6,7,8,9,10,11,12,13,14,15);

        v16bf bfrag[4];
#pragma unroll
        for (int nt = 0; nt < 4; ++nt) {
            const __bf16* bp = w2t + (size_t)(nt * 16 + l15) * N1 + kb + koff;
            v8bf blo = *(const v8bf*)(bp);
            v8bf bhi = *(const v8bf*)(bp + 16);
            bfrag[nt] = __builtin_shufflevector(blo, bhi,
                0,1,2,3,4,5,6,7,8,9,10,11,12,13,14,15);
        }
#pragma unroll
        for (int nt = 0; nt < 4; ++nt)
            acc[nt] = __builtin_amdgcn_wmma_f32_16x16x32_bf16(
                false, afrag, false, bfrag[nt],
                (short)0, acc[nt], false, false);
    }

#pragma unroll
    for (int nt = 0; nt < 4; ++nt) {
        const int col = nt * 16 + l15;
        if (col < N2) {
            const float bias = b2[col];
#pragma unroll
            for (int v = 0; v < 8; ++v) {
                const int row = rowBase + ((lane < 16) ? v : v + 8);
                out[(size_t)row * N2 + col] = acc[nt][v] + bias;
            }
        }
    }
}

extern "C" void kernel_launch(void* const* d_in, const int* in_sizes, int n_in,
                              void* d_out, int out_size, void* d_ws, size_t ws_size,
                              hipStream_t stream) {
    const float* x  = (const float*)d_in[0];
    const float* w1 = (const float*)d_in[1];
    const float* b1 = (const float*)d_in[2];
    const float* w2 = (const float*)d_in[3];
    const float* b2 = (const float*)d_in[4];
    float* out = (float*)d_out;

    char* ws = (char*)d_ws;
    __bf16* w1t = (__bf16*)ws;                                   // N1*K1 bf16 = 6 MB
    __bf16* w2t = (__bf16*)(ws + (size_t)N1 * K1 * 2);           // N2P*N1 bf16 = 128 KB
    __bf16* h   = (__bf16*)(ws + (size_t)N1 * K1 * 2
                               + (size_t)N2P * N1 * 2);          // RROWS*N1 bf16 = 64 MB

    pack_w1t<<<(K1 * N1) / 256, 256, 0, stream>>>(w1, w1t);
    pack_w2t<<<(N2P * N1) / 256, 256, 0, stream>>>(w2, w2t);

    dim3 g1(N1 / 128, RROWS / 128);
    gemm1_relu_tdm<<<g1, 256, 0, stream>>>(x, w1t, b1, h);
    gemm2_bias<<<RROWS / 128, 256, 0, stream>>>(h, w2t, b2, out);
}
